// LatexDecoder_89781996355824
// MI455X (gfx1250) — matrix operand order repaired and checked by reading, not compiled, not run on hardware
//
#include <hip/hip_runtime.h>
#include <math.h>

// ---------------------------------------------------------------------------
//   V=2000 vocab, E=256 embed, Hd=256 hidden, ENC=256 channels, A=256 attn
//   B=64 batch, HI=16, WI=32 -> P=512 positions, T=256 steps
// ---------------------------------------------------------------------------
#define B_   64
#define T_   256
#define V_   2000
#define E_   256
#define HD_  256
#define ENC_ 256
#define A_   256
#define P_   512

typedef __attribute__((ext_vector_type(16))) __bf16 v16bf;
typedef __attribute__((ext_vector_type(8)))  float  v8f;

__device__ __forceinline__ float sigmoidf_(float v) {
    return 1.0f / (1.0f + __expf(-v));
}

#define WMMA_BF16(Af, Bf, Cacc) \
    __builtin_amdgcn_wmma_f32_16x16x32_bf16(false, (Af), false, (Bf), (short)0, (Cacc), false, false)

// ---------------------------------------------------------------------------
// A-fragment builders (16x32 bf16, ISA 7.12.2):
//   lane half h=lane>>4, elem j: K = (j<8 ? j : j+8) + 8h, M-row = lane&15
//   -> two contiguous 8-float runs per lane when K-stride==1 (float4 loads)
// ---------------------------------------------------------------------------
__device__ __forceinline__ v16bf frag_a_vec(const float* __restrict__ rowp,
                                            int k0, int half)
{
    const float* p = rowp + k0 + (half << 3);
    const float4 a0 = *(const float4*)(p);
    const float4 a1 = *(const float4*)(p + 4);
    const float4 a2 = *(const float4*)(p + 16);
    const float4 a3 = *(const float4*)(p + 20);
    v16bf f;
    f[0]=(__bf16)a0.x;  f[1]=(__bf16)a0.y;  f[2]=(__bf16)a0.z;  f[3]=(__bf16)a0.w;
    f[4]=(__bf16)a1.x;  f[5]=(__bf16)a1.y;  f[6]=(__bf16)a1.z;  f[7]=(__bf16)a1.w;
    f[8]=(__bf16)a2.x;  f[9]=(__bf16)a2.y;  f[10]=(__bf16)a2.z; f[11]=(__bf16)a2.w;
    f[12]=(__bf16)a3.x; f[13]=(__bf16)a3.y; f[14]=(__bf16)a3.z; f[15]=(__bf16)a3.w;
    return f;
}

__device__ __forceinline__ v16bf frag_a_gen(const float* __restrict__ base,
                                            long arow, long sAk, int k0, int half)
{
    v16bf f;
#pragma unroll
    for (int j = 0; j < 16; ++j) {
        const int ka = k0 + ((j < 8) ? j : (j + 8)) + (half << 3);
        f[j] = (__bf16)base[arow + (long)ka * sAk];
    }
    return f;
}

// ---------------------------------------------------------------------------
// Weight pre-pack: W[N,K] f32 (row-major)  ->  bf16 fragments laid out as
//   [N/16 tiles][K/32 steps][32 lanes][16 bf16]   (32 bytes per lane-fragment)
// B-fragment layout (32x16): elem j -> K = kt*32 + 8h + (j<8 ? j : j+8), N-col = nt*16+(lane&15)
// One thread per (tile,kstep,lane): 16 cvt + one 32B store. Runs once.
// ---------------------------------------------------------------------------
__global__ void k_pack_wT(const float* __restrict__ W, int N, int K, int total,
                          __bf16* __restrict__ out)
{
    const int gid = blockIdx.x * blockDim.x + threadIdx.x;
    if (gid >= total) return;
    const int lane = gid & 31;
    const int fk   = gid >> 5;          // nt * (K/32) + kt
    const int nkt  = K >> 5;
    const int nt   = fk / nkt;
    const int kt   = fk - nt * nkt;
    const float* src = W + (long)(nt * 16 + (lane & 15)) * K
                         + kt * 32 + ((lane >> 4) << 3);
    __bf16* dst = out + (long)gid * 16;
#pragma unroll
    for (int j = 0; j < 16; ++j) dst[j] = (__bf16)src[(j < 8) ? j : (j + 8)];
}

__device__ __forceinline__ void store_tile(float* __restrict__ Cb, long ldC,
                                           int m0, int ntile_base, int idx16, int half,
                                           v8f acc,
                                           const float* __restrict__ bias0,
                                           const float* __restrict__ bias1, int act)
{
    const int nn = ntile_base + idx16;
    float badd = 0.0f;
    if (bias0) badd += bias0[nn];
    if (bias1) badd += bias1[nn];
    const int mb = m0 + (half << 3);
#pragma unroll
    for (int r = 0; r < 8; ++r) {
        float v = acc[r] + badd;
        if (act == 1) v = tanhf(v);
        Cb[(long)(mb + r) * ldC + nn] = v;
    }
}

// ---------------------------------------------------------------------------
// Generic  C[M,N] = act( X[M,K] @ W[N,K]^T + bias0 + bias1 )
// W is pre-packed bf16 fragments: each fragment is ONE aligned 32B load/lane,
// zero conversions. One wave does a 16x64 strip (A reused over 4 B tiles),
// 4 WMMA per K step. blockIdx.z batches independent GEMMs (att1).
// ---------------------------------------------------------------------------
__global__ void wmma_gemm_xwT(const float* __restrict__ Ap, long sAm, long sAk, long sAb,
                              const __bf16* __restrict__ Wp, int nkt,  // K/32
                              float* __restrict__ Cp, long ldC, long sCb,
                              const float* __restrict__ bias0,
                              const float* __restrict__ bias1,
                              int M, int N, int act)
{
    const int lane = threadIdx.x;          // 32 threads = 1 wave
    const int nt0 = blockIdx.x << 2;       // 4 N-tiles of 16
    const int n0  = nt0 << 4;
    const int m0  = blockIdx.y << 4;
    if (m0 + 16 > M) return;

    const float* Ab = Ap + (long)blockIdx.z * sAb;
    float*       Cb = Cp + (long)blockIdx.z * sCb;

    const int idx16 = lane & 15;
    const int half  = lane >> 4;
    const long arow = (long)(m0 + idx16) * sAm;
    const float* arp = Ab + arow;

    const bool t1 = (n0 + 32) <= N;
    const bool t2 = (n0 + 48) <= N;
    const bool t3 = (n0 + 64) <= N;

    const v16bf* wb = (const v16bf*)Wp;
    const v16bf* f0 = wb + ((long)(nt0 + 0) * nkt) * 32 + lane;
    const v16bf* f1 = wb + ((long)(nt0 + 1) * nkt) * 32 + lane;
    const v16bf* f2 = wb + ((long)(nt0 + 2) * nkt) * 32 + lane;
    const v16bf* f3 = wb + ((long)(nt0 + 3) * nkt) * 32 + lane;

    v8f a0 = {}, a1 = {}, a2 = {}, a3 = {};

    if (sAk == 1) {
        for (int kt = 0; kt < nkt; ++kt) {
            const v16bf af = frag_a_vec(arp, kt << 5, half);
            a0 = WMMA_BF16(af, f0[kt * 32], a0);
            if (t1) a1 = WMMA_BF16(af, f1[kt * 32], a1);
            if (t2) a2 = WMMA_BF16(af, f2[kt * 32], a2);
            if (t3) a3 = WMMA_BF16(af, f3[kt * 32], a3);
        }
    } else {
        for (int kt = 0; kt < nkt; ++kt) {
            const v16bf af = frag_a_gen(Ab, arow, sAk, kt << 5, half);
            a0 = WMMA_BF16(af, f0[kt * 32], a0);
            if (t1) a1 = WMMA_BF16(af, f1[kt * 32], a1);
            if (t2) a2 = WMMA_BF16(af, f2[kt * 32], a2);
            if (t3) a3 = WMMA_BF16(af, f3[kt * 32], a3);
        }
    }

    store_tile(Cb, ldC, m0, n0,      idx16, half, a0, bias0, bias1, act);
    if (t1) store_tile(Cb, ldC, m0, n0 + 16, idx16, half, a1, bias0, bias1, act);
    if (t2) store_tile(Cb, ldC, m0, n0 + 32, idx16, half, a2, bias0, bias1, act);
    if (t3) store_tile(Cb, ldC, m0, n0 + 48, idx16, half, a3, bias0, bias1, act);
}

// ---------------------------------------------------------------------------
// Fully fused LSTM layer: gates = X@Wih^T + Hprev@Whh^T + biases, then the
// cell — all in registers. A wave owns 16 batch rows x 32 hidden columns and
// holds the i/f/g/o accumulators for those SAME columns, so the four gate
// tiles are element-aligned in the C/D layout: cell runs with zero LDS.
// Weights come pre-packed (one 32B load per fragment). 8 WMMA per K step.
// Grid: (HD_/32, B_/16) x 32 threads.
// ---------------------------------------------------------------------------
__global__ void k_lstm_layer(const float* __restrict__ X, int ldX, int nkt1,
                             const __bf16* __restrict__ WihP,   // packed [4Hd, K1]
                             const float* __restrict__ Hprev,   // [B, Hd]
                             const __bf16* __restrict__ WhhP,   // packed [4Hd, Hd]
                             const float* __restrict__ bih,
                             const float* __restrict__ bhh,
                             const float* __restrict__ Cin,
                             float* __restrict__ Hout,
                             float* __restrict__ Cout)
{
    const int lane  = threadIdx.x;
    const int col0  = blockIdx.x << 5;   // 32 hidden columns
    const int m0    = blockIdx.y << 4;   // 16 batch rows
    const int idx16 = lane & 15;
    const int half  = lane >> 4;

    v8f acc[8];
#pragma unroll
    for (int q = 0; q < 8; ++q) acc[q] = (v8f){};

    const v16bf* wfrag[8];

    // pass 1: X @ Wih^T
#pragma unroll
    for (int g = 0; g < 4; ++g)
#pragma unroll
        for (int s = 0; s < 2; ++s) {
            const int nt = g * (HD_ / 16) + (col0 >> 4) + s;
            wfrag[g * 2 + s] = (const v16bf*)WihP + ((long)nt * nkt1) * 32 + lane;
        }
    const float* xr = X + (long)(m0 + idx16) * ldX;
    for (int kt = 0; kt < nkt1; ++kt) {
        const v16bf af = frag_a_vec(xr, kt << 5, half);
#pragma unroll
        for (int q = 0; q < 8; ++q)
            acc[q] = WMMA_BF16(af, wfrag[q][kt * 32], acc[q]);
    }

    // pass 2: Hprev @ Whh^T   (K = Hd, nkt = Hd/32)
#pragma unroll
    for (int g = 0; g < 4; ++g)
#pragma unroll
        for (int s = 0; s < 2; ++s) {
            const int nt = g * (HD_ / 16) + (col0 >> 4) + s;
            wfrag[g * 2 + s] = (const v16bf*)WhhP + ((long)nt * (HD_ / 32)) * 32 + lane;
        }
    const float* hr = Hprev + (long)(m0 + idx16) * HD_;
    for (int kt = 0; kt < HD_ / 32; ++kt) {
        const v16bf af = frag_a_vec(hr, kt << 5, half);
#pragma unroll
        for (int q = 0; q < 8; ++q)
            acc[q] = WMMA_BF16(af, wfrag[q][kt * 32], acc[q]);
    }

    // cell epilogue, fully in registers (PyTorch gate order i,f,g,o)
#pragma unroll
    for (int s = 0; s < 2; ++s) {
        const int col = col0 + s * 16 + idx16;
        const float bi = bih[col]           + bhh[col];
        const float bf = bih[HD_ + col]     + bhh[HD_ + col];
        const float bg = bih[2 * HD_ + col] + bhh[2 * HD_ + col];
        const float bo = bih[3 * HD_ + col] + bhh[3 * HD_ + col];
#pragma unroll
        for (int r = 0; r < 8; ++r) {
            const int  m   = m0 + (half << 3) + r;
            const long idx = (long)m * HD_ + col;
            const float gi = acc[0 + s][r] + bi;
            const float gf = acc[2 + s][r] + bf;
            const float gg = acc[4 + s][r] + bg;
            const float go = acc[6 + s][r] + bo;
            const float c2 = sigmoidf_(gf) * Cin[idx] + sigmoidf_(gi) * tanhf(gg);
            Cout[idx] = c2;
            Hout[idx] = sigmoidf_(go) * tanhf(c2);
        }
    }
}

// ---------------------------------------------------------------------------
// Spatial mean: wave-per-(b,c)-row, coalesced b32 lane reads + shuffle reduce.
// ---------------------------------------------------------------------------
__global__ void k_mean(const float* __restrict__ enc, float* __restrict__ mean)
{
    const int b = blockIdx.x, tid = threadIdx.x;
    const int wid = tid >> 5, lane = tid & 31;
    for (int c = wid; c < ENC_; c += 8) {
        const float* p = enc + ((long)b * ENC_ + c) * P_;
        float s = 0.0f;
#pragma unroll 4
        for (int i = lane; i < P_; i += 32) s += p[i];
#pragma unroll
        for (int m = 16; m > 0; m >>= 1) s += __shfl_xor(s, m, 32);
        if (lane == 0) mean[b * ENC_ + c] = s * (1.0f / (float)P_);
    }
}

__global__ void k_dup(const float* __restrict__ src, float* __restrict__ dst)
{
    const int i = blockIdx.x * blockDim.x + threadIdx.x;
    dst[i] = src[i];
}

// ---------------------------------------------------------------------------
// Fused attention step (block per batch row, 256 threads = 8 waves).
// All long-row reductions are wave-cooperative: 32 lanes read CONTIGUOUS
// 32B chunks (coalesced global_load_b128) and reduce with __shfl_xor.
//   att2[a] = h1[b,:].W_ad[a,:] + b_ad[a]
//   score[p] = b_fa + sum_a W_fa[a]*tanh(att1[b,p,a]+att2[a])
//   w = softmax(score) -> d_out attention section
//   ctx[c] = sum_p w[p]*encoder_out[b,c,p]
//   x[b]   = concat(emb[captions[b,t]], ctx)
// att1 + enc (66MB/step) are L2-resident on MI455X (192MB L2).
// ---------------------------------------------------------------------------
__global__ void k_score_softmax_ctx(const float* __restrict__ att1,
                                    const float* __restrict__ h1,
                                    const float* __restrict__ W_ad,
                                    const float* __restrict__ b_ad,
                                    const float* __restrict__ W_fa,
                                    const float* __restrict__ b_fa,
                                    const float* __restrict__ enc,
                                    const float* __restrict__ emb,
                                    const int*   __restrict__ caps,
                                    int t,
                                    float* __restrict__ w_out,
                                    float* __restrict__ x)
{
    __shared__ float s_h[HD_];
    __shared__ float s_att2[A_];
    __shared__ float s_wfa[A_];
    __shared__ float s_sc[P_];
    __shared__ float red[256];

    const int b = blockIdx.x, tid = threadIdx.x;
    const int wid = tid >> 5, lane = tid & 31;

    s_h[tid]   = h1[b * HD_ + tid];
    s_wfa[tid] = W_fa[tid];
    __syncthreads();

    // att2: wave per attention row, coalesced float4 pairs + shuffle reduce
    for (int a = wid; a < A_; a += 8) {
        const float* wr = W_ad + (long)a * HD_ + lane * 8;
        const float4 w0 = *(const float4*)wr;
        const float4 w1 = *(const float4*)(wr + 4);
        const int k0 = lane * 8;
        float s =          w0.x * s_h[k0 + 0];
        s = fmaf(w0.y, s_h[k0 + 1], s);
        s = fmaf(w0.z, s_h[k0 + 2], s);
        s = fmaf(w0.w, s_h[k0 + 3], s);
        s = fmaf(w1.x, s_h[k0 + 4], s);
        s = fmaf(w1.y, s_h[k0 + 5], s);
        s = fmaf(w1.z, s_h[k0 + 6], s);
        s = fmaf(w1.w, s_h[k0 + 7], s);
#pragma unroll
        for (int m = 16; m > 0; m >>= 1) s += __shfl_xor(s, m, 32);
        if (lane == 0) s_att2[a] = s + b_ad[a];
    }
    __syncthreads();

    // score: wave per position row; lane reads 32B contiguous of att1 row
    const float bfa = b_fa[0];
    for (int p = wid; p < P_; p += 8) {
        const float* row = att1 + ((long)b * P_ + p) * A_ + lane * 8;
        __builtin_prefetch(row + 8 * A_, 0, 0);
        const float4 r0 = *(const float4*)row;
        const float4 r1 = *(const float4*)(row + 4);
        const int a0 = lane * 8;
        float s =          s_wfa[a0 + 0] * tanhf(r0.x + s_att2[a0 + 0]);
        s = fmaf(s_wfa[a0 + 1], tanhf(r0.y + s_att2[a0 + 1]), s);
        s = fmaf(s_wfa[a0 + 2], tanhf(r0.z + s_att2[a0 + 2]), s);
        s = fmaf(s_wfa[a0 + 3], tanhf(r0.w + s_att2[a0 + 3]), s);
        s = fmaf(s_wfa[a0 + 4], tanhf(r1.x + s_att2[a0 + 4]), s);
        s = fmaf(s_wfa[a0 + 5], tanhf(r1.y + s_att2[a0 + 5]), s);
        s = fmaf(s_wfa[a0 + 6], tanhf(r1.z + s_att2[a0 + 6]), s);
        s = fmaf(s_wfa[a0 + 7], tanhf(r1.w + s_att2[a0 + 7]), s);
#pragma unroll
        for (int m = 16; m > 0; m >>= 1) s += __shfl_xor(s, m, 32);
        if (lane == 0) s_sc[p] = s + bfa;
    }
    __syncthreads();

    // softmax over 512 positions (block reduction)
    const float v0 = s_sc[tid], v1 = s_sc[tid + 256];
    red[tid] = fmaxf(v0, v1);
    __syncthreads();
    for (int s = 128; s > 0; s >>= 1) {
        if (tid < s) red[tid] = fmaxf(red[tid], red[tid + s]);
        __syncthreads();
    }
    const float mx = red[0];
    __syncthreads();

    float e0 = __expf(v0 - mx), e1 = __expf(v1 - mx);
    red[tid] = e0 + e1;
    __syncthreads();
    for (int s = 128; s > 0; s >>= 1) {
        if (tid < s) red[tid] += red[tid + s];
        __syncthreads();
    }
    const float inv = 1.0f / red[0];
    __syncthreads();

    e0 *= inv; e1 *= inv;
    s_sc[tid] = e0; s_sc[tid + 256] = e1;
    float* wo = w_out + ((long)b * T_ + t) * P_;
    wo[tid] = e0; wo[tid + 256] = e1;
    __syncthreads();

    // ctx: wave per channel; lanes stride the position axis (coalesced)
    for (int c = wid; c < ENC_; c += 8) {
        const float* er = enc + ((long)b * ENC_ + c) * P_;
        float s = 0.0f;
#pragma unroll 4
        for (int p = lane; p < P_; p += 32) s = fmaf(s_sc[p], er[p], s);
#pragma unroll
        for (int m = 16; m > 0; m >>= 1) s += __shfl_xor(s, m, 32);
        if (lane == 0) x[(long)b * (E_ + ENC_) + E_ + c] = s;
    }

    // embedding half of x
    const int tok = caps[b * T_ + t];
    x[(long)b * (E_ + ENC_) + tid] = emb[(long)tok * E_ + tid];
}

// ---------------------------------------------------------------------------
extern "C" void kernel_launch(void* const* d_in, const int* in_sizes, int n_in,
                              void* d_out, int out_size, void* d_ws, size_t ws_size,
                              hipStream_t stream)
{
    const float* enc     = (const float*)d_in[0];
    const int*   caps    = (const int*)  d_in[1];
    const float* emb     = (const float*)d_in[2];
    const float* W_ae    = (const float*)d_in[3];
    const float* b_ae    = (const float*)d_in[4];
    const float* W_ad    = (const float*)d_in[5];
    const float* b_ad    = (const float*)d_in[6];
    const float* W_fa    = (const float*)d_in[7];
    const float* b_fa    = (const float*)d_in[8];
    const float* W_inith = (const float*)d_in[9];
    const float* b_inith = (const float*)d_in[10];
    const float* W_initc = (const float*)d_in[11];
    const float* b_initc = (const float*)d_in[12];
    const float* W_ih0   = (const float*)d_in[13];
    const float* W_hh0   = (const float*)d_in[14];
    const float* b_ih0   = (const float*)d_in[15];
    const float* b_hh0   = (const float*)d_in[16];
    const float* W_ih1   = (const float*)d_in[17];
    const float* W_hh1   = (const float*)d_in[18];
    const float* b_ih1   = (const float*)d_in[19];
    const float* b_hh1   = (const float*)d_in[20];
    const float* W_out   = (const float*)d_in[21];
    const float* b_out   = (const float*)d_in[22];

    // f32 workspace
    float* ws    = (float*)d_ws;
    float* mean  = ws; ws += B_ * ENC_;
    float* att1  = ws; ws += (long)B_ * P_ * A_;
    float* h0a   = ws; ws += B_ * HD_;
    float* h0b   = ws; ws += B_ * HD_;
    float* h1a   = ws; ws += B_ * HD_;
    float* h1b   = ws; ws += B_ * HD_;
    float* c0    = ws; ws += B_ * HD_;
    float* c1    = ws; ws += B_ * HD_;
    float* x     = ws; ws += B_ * (E_ + ENC_);

    // packed bf16 weight fragments (carved from same workspace; 2 bf16/float)
    __bf16* pWae  = (__bf16*)ws; ws += (A_ * ENC_) / 2;
    __bf16* pWih  = (__bf16*)ws; ws += (4 * HD_ * HD_) / 2;          // W_inith+W_initc share? no:
    __bf16* pWic  = (__bf16*)ws; ws += (HD_ * ENC_) / 2;
    __bf16* pWih0 = (__bf16*)ws; ws += (4 * HD_ * (E_ + ENC_)) / 2;
    __bf16* pWhh0 = (__bf16*)ws; ws += (4 * HD_ * HD_) / 2;
    __bf16* pWih1 = (__bf16*)ws; ws += (4 * HD_ * HD_) / 2;
    __bf16* pWhh1 = (__bf16*)ws; ws += (4 * HD_ * HD_) / 2;
    __bf16* pWout = (__bf16*)ws; ws += (V_ * HD_) / 2;
    // note: pWih above is the packed W_inith (HD_*ENC_ elems) -- it was carved
    // with extra room (4*HD_*HD_/2 floats), harmless over-allocation.

    float* outv = (float*)d_out;                 // [B,T,V]
    float* outw = outv + (long)B_ * T_ * V_;     // [B,T,P]

    // ---- one-time: pack all GEMM weights to bf16 fragment layout --------
    {
        int th;
        th = (A_ * ENC_) / 16;
        k_pack_wT<<<(th + 255) / 256, 256, 0, stream>>>(W_ae, A_, ENC_, th, pWae);
        th = (HD_ * ENC_) / 16;
        k_pack_wT<<<(th + 255) / 256, 256, 0, stream>>>(W_inith, HD_, ENC_, th, pWih);
        k_pack_wT<<<(th + 255) / 256, 256, 0, stream>>>(W_initc, HD_, ENC_, th, pWic);
        th = (4 * HD_ * (E_ + ENC_)) / 16;
        k_pack_wT<<<(th + 255) / 256, 256, 0, stream>>>(W_ih0, 4 * HD_, E_ + ENC_, th, pWih0);
        th = (4 * HD_ * HD_) / 16;
        k_pack_wT<<<(th + 255) / 256, 256, 0, stream>>>(W_hh0, 4 * HD_, HD_, th, pWhh0);
        k_pack_wT<<<(th + 255) / 256, 256, 0, stream>>>(W_ih1, 4 * HD_, HD_, th, pWih1);
        k_pack_wT<<<(th + 255) / 256, 256, 0, stream>>>(W_hh1, 4 * HD_, HD_, th, pWhh1);
        th = (V_ * HD_) / 16;
        k_pack_wT<<<(th + 255) / 256, 256, 0, stream>>>(W_out, V_, HD_, th, pWout);
    }

    // ---- one-time setup -------------------------------------------------
    k_mean<<<B_, 256, 0, stream>>>(enc, mean);

    wmma_gemm_xwT<<<dim3(HD_ / 64, B_ / 16, 1), 32, 0, stream>>>(
        mean, ENC_, 1, 0, pWih, ENC_ / 32, h0a, HD_, 0, b_inith, nullptr,
        B_, HD_, /*act=*/1);
    wmma_gemm_xwT<<<dim3(HD_ / 64, B_ / 16, 1), 32, 0, stream>>>(
        mean, ENC_, 1, 0, pWic, ENC_ / 32, c0, HD_, 0, b_initc, nullptr,
        B_, HD_, 1);
    k_dup<<<(B_ * HD_) / 256, 256, 0, stream>>>(h0a, h1a);
    k_dup<<<(B_ * HD_) / 256, 256, 0, stream>>>(c0, c1);

    // att1[b,p,a] = enc[b,p,:] @ W_ae^T + b_ae  (strided A: sAm=1, sAk=P)
    wmma_gemm_xwT<<<dim3(A_ / 64, P_ / 16, B_), 32, 0, stream>>>(
        enc, 1, P_, (long)ENC_ * P_, pWae, ENC_ / 32,
        att1, A_, (long)P_ * A_, b_ae, nullptr,
        P_, A_, 0);

    // ---- sequential scan: 4 kernels per step ---------------------------
    float* h0c = h0a; float* h0n = h0b;
    float* h1c = h1a; float* h1n = h1b;
    for (int t = 0; t < T_; ++t) {
        k_score_softmax_ctx<<<B_, 256, 0, stream>>>(
            att1, h1c, W_ad, b_ad, W_fa, b_fa, enc, emb, caps, t, outw, x);

        k_lstm_layer<<<dim3(HD_ / 32, B_ / 16), 32, 0, stream>>>(
            x, E_ + ENC_, (E_ + ENC_) / 32, pWih0, h0c, pWhh0, b_ih0, b_hh0,
            c0, h0n, c0);

        k_lstm_layer<<<dim3(HD_ / 32, B_ / 16), 32, 0, stream>>>(
            h0n, HD_, HD_ / 32, pWih1, h1c, pWhh1, b_ih1, b_hh1,
            c1, h1n, c1);

        wmma_gemm_xwT<<<dim3((V_ + 63) / 64, B_ / 16, 1), 32, 0, stream>>>(
            h1n, HD_, 1, 0, pWout, HD_ / 32, outv + (long)t * V_, (long)T_ * V_, 0,
            b_out, nullptr, B_, V_, 0);

        float* tmp;
        tmp = h0c; h0c = h0n; h0n = tmp;
        tmp = h1c; h1c = h1n; h1n = tmp;
    }
}
